// model_83769042141391
// MI455X (gfx1250) — matrix operand order repaired and checked by reading, not compiled, not run on hardware
//
#include <hip/hip_runtime.h>
#include <hip/hip_bf16.h>
#include <math.h>

// ---------------- problem constants ----------------
constexpr int kB   = 64;     // batch
constexpr int kS   = 512;    // seq len
constexpr int kE   = 512;    // embed dim
constexpr int kH   = 256;    // per-direction hidden
constexpr int kG   = 1024;   // 4*H gate width
constexpr int kHID = 512;    // bilstm output dim
constexpr int kATT = 256;
constexpr int kC   = 10;

typedef __attribute__((ext_vector_type(16))) _Float16 v16h;
typedef __attribute__((ext_vector_type(8)))  _Float16 v8h;
typedef __attribute__((ext_vector_type(8)))  float    v8f;
typedef int v4i __attribute__((vector_size(16)));

// ---- gfx1250 async global->LDS copy (guarded; falls back to plain copy) ---
#if defined(__has_builtin)
#if __has_builtin(__builtin_amdgcn_global_load_async_to_lds_b128)
#define HAVE_ASYNC_LDS 1
#endif
#endif
#ifndef HAVE_ASYNC_LDS
#define HAVE_ASYNC_LDS 0
#endif

__device__ inline void ldsCopy8(_Float16* d, const _Float16* s) {
#if HAVE_ASYNC_LDS
  __builtin_amdgcn_global_load_async_to_lds_b128(
      (__attribute__((address_space(1))) v4i*)s,
      (__attribute__((address_space(3))) v4i*)d, 0, 0);
#else
  *(v8h*)d = *(const v8h*)s;
#endif
}

__device__ inline void asyncWait() {
#if HAVE_ASYNC_LDS
#if __has_builtin(__builtin_amdgcn_s_wait_asynccnt)
  __builtin_amdgcn_s_wait_asynccnt(0);
#else
  asm volatile("s_wait_asynccnt 0x0" ::: "memory");
#endif
#endif
}

// Stage nkt chunks of 2048 halves (4 packed B tiles each) into LDS.
// src chunk k lives at src + k*ktStride; dst is contiguous. 256 threads.
__device__ inline void stage4(_Float16* __restrict__ dstLds,
                              const _Float16* __restrict__ src,
                              int nkt, size_t ktStride, int tid) {
  for (int c = tid; c < nkt * 256; c += 256) {
    int kt = c >> 8;
    int w  = (c & 255) * 8;
    ldsCopy8(dstLds + ((size_t)kt << 11) + w, src + (size_t)kt * ktStride + w);
  }
}

__device__ inline v8f wmma_f16(v16h a, v16h b, v8f c) {
  // D = A(16x32 f16) x B(32x16 f16) + C(16x16 f32)
  return __builtin_amdgcn_wmma_f32_16x16x32_f16(
      /*neg_a=*/false, a, /*neg_b=*/false, b,
      /*c_mod=*/(short)0, c, /*reuse_a=*/false, /*reuse_b=*/false);
}

// A fragment (16x32 f16, row-major source, lda in halves).
// ISA layout: lane L holds row m=L&15; halves 0..7 <- k = kb..kb+7,
// halves 8..15 <- k = kb+16..kb+23, where kb = ktBase + (L>=16 ? 8 : 0).
__device__ inline v16h loadAfragH(const _Float16* __restrict__ base, int lda,
                                  int m, int kb) {
  union { v16h v; v8h h[2]; } a;
  const _Float16* r = base + (size_t)m * lda + kb;
  a.h[0] = *(const v8h*)(r);
  a.h[1] = *(const v8h*)(r + 16);
  return a.v;
}

// WMMA C-layout fragment <-> per-lane-contiguous memory (producer/consumer
// symmetric: layout-assumption free for handoff buffers).
__device__ inline void storeCfrag(float* dst, v8f acc) {
  union { v8f v; float4 f[2]; } u; u.v = acc;
  *(float4*)dst       = u.f[0];
  *(float4*)(dst + 4) = u.f[1];
}
__device__ inline v8f loadCfrag(const float* src) {
  union { v8f v; float4 f[2]; } u;
  u.f[0] = *(const float4*)src;
  u.f[1] = *(const float4*)(src + 4);
  return u.v;
}

__device__ inline float sigm(float x) { return 1.f / (1.f + __expf(-x)); }

// ---------------- 1) embedding gather -> f16 time-major [S*B][E] ----------
__global__ void k_gather(const int* __restrict__ x, const float* __restrict__ emb,
                         _Float16* __restrict__ embH) {
  size_t tid  = (size_t)blockIdx.x * blockDim.x + threadIdx.x;
  size_t base = tid * 8;                       // 8 elements per thread
  int    e    = (int)(base % kE);
  size_t sb   = base / kE;                     // sb = s*B + b (time-major)
  int    s    = (int)(sb / kB);
  int    b    = (int)(sb % kB);
  int    tok  = x[b * kS + s];
  const float* src = emb + (size_t)tok * kE + e;
  float4 f0 = *(const float4*)(src);
  float4 f1 = *(const float4*)(src + 4);
  v8h o;
  o[0]=(_Float16)f0.x; o[1]=(_Float16)f0.y; o[2]=(_Float16)f0.z; o[3]=(_Float16)f0.w;
  o[4]=(_Float16)f1.x; o[5]=(_Float16)f1.y; o[6]=(_Float16)f1.z; o[7]=(_Float16)f1.w;
  *(v8h*)(embH + sb * kE + e) = o;
}

// ------ 2) pack W [N rows][K cols] f32 into WMMA-B fragment order f16 ------
__global__ void k_pack(const float* __restrict__ W, _Float16* __restrict__ out,
                       int K, int Ntiles) {
  size_t p    = (size_t)blockIdx.x * blockDim.x + threadIdx.x;
  int    i    = (int)(p & 15);
  int    lane = (int)((p >> 4) & 31);
  size_t tile = p >> 9;
  int    nt   = (int)(tile % Ntiles);
  int    kt   = (int)(tile / Ntiles);
  int    n    = nt * 16 + (lane & 15);
  int    k    = kt * 32 + ((lane & 16) ? 8 : 0) + (i & 7) + ((i & 8) ? 16 : 0);
  out[p] = (_Float16)W[(size_t)n * K + k];
}

// ---------------- 3) combined gate bias ----------------
__global__ void k_bias(const float* __restrict__ bi, const float* __restrict__ bh,
                       float* __restrict__ out) {
  int g = blockIdx.x * blockDim.x + threadIdx.x;   // 1024
  out[g] = bi[g] + bh[g];
}

// ------ 4) gx = embH(32768x512) @ wihT(512x1024) + bias  (per dir) --------
// Block = 8 row tiles x one ntb (4 col tiles) x dir. The 4 B column tiles
// (64 KB packed f16) are staged into LDS once via async global->LDS loads and
// shared by all 8 waves. Output in WMMA C-fragment order for the LSTM kernel.
__global__ __launch_bounds__(256) void k_gemm_gx(const _Float16* __restrict__ embH,
                                                 const _Float16* __restrict__ wihP,
                                                 const float* __restrict__ biasC,
                                                 float* __restrict__ gxP) {
  __shared__ _Float16 bsh[16 * 2048];   // [kt][t][lane][16] = 64 KB
  int dir = blockIdx.z;
  int ntb = blockIdx.y;                  // 0..15
  const _Float16* bp   = wihP  + (size_t)dir * (kE * kG);
  const float*    bias = biasC + dir * kG;
  float*          gxD  = gxP   + (size_t)dir * ((size_t)kS * kB * kG);

  int tid = threadIdx.x;
  stage4(bsh, bp + (size_t)ntb * 2048, 16, 64 * 512, tid);
  asyncWait();
  __syncthreads();

  int lane   = tid & 31;
  int laneHi = (lane >> 4) & 1;
  int mt     = blockIdx.x * 8 + (tid >> 5);   // 2048 row tiles
  int m      = mt * 16 + (lane & 15);

  v8f acc[4];
#pragma unroll
  for (int t = 0; t < 4; ++t) {
    float bv = bias[(ntb * 4 + t) * 16 + (lane & 15)];
#pragma unroll
    for (int v = 0; v < 8; ++v) acc[t][v] = bv;
  }

  const _Float16* rowA = embH + (size_t)m * kE;
#pragma unroll 2
  for (int kt = 0; kt < kE / 32; ++kt) {
    int kb = kt * 32 + laneHi * 8;
    union { v16h v; v8h h[2]; } a;
    a.h[0] = *(const v8h*)(rowA + kb);
    a.h[1] = *(const v8h*)(rowA + kb + 16);
    __builtin_prefetch(rowA + kb + 64, 0, 3);   // global_prefetch_b8, near
#pragma unroll
    for (int t = 0; t < 4; ++t) {
      v16h b = *(const v16h*)(bsh + ((kt * 4 + t) * 512 + lane * 16));
      acc[t] = wmma_f16(a.v, b, acc[t]);
    }
  }
#pragma unroll
  for (int t = 0; t < 4; ++t) {
    int nt = ntb * 4 + t;
    storeCfrag(gxD + (((size_t)mt * 64 + nt) * 32 + lane) * 8, acc[t]);
  }
}

// ------ 5) LSTM recurrence: one persistent workgroup per direction --------
__global__ __launch_bounds__(1024) void k_lstm(const float* __restrict__ gxP,
                                               const _Float16* __restrict__ whhP,
                                               float* __restrict__ lstm,
                                               _Float16* __restrict__ lstmH) {
  __shared__ _Float16 hbuf[kB * kH];   // 32 KB, f16 hidden state
  int dir = blockIdx.x;
  const float*    gxD = gxP  + (size_t)dir * ((size_t)kS * kB * kG);
  const _Float16* bp  = whhP + (size_t)dir * (kH * kG);

  int tid = threadIdx.x;
  int wave = tid >> 5, lane = tid & 31, laneHi = (lane >> 4) & 1;

  for (int i = tid; i < kB * kH; i += 1024) hbuf[i] = (_Float16)0.f;

  float    cst[2][8];
  _Float16 hh[2][8];
#pragma unroll
  for (int q = 0; q < 2; ++q)
#pragma unroll
    for (int v = 0; v < 8; ++v) cst[q][v] = 0.f;

  __syncthreads();

  for (int step = 0; step < kS; ++step) {
    int s = dir ? (kS - 1 - step) : step;

#pragma unroll
    for (int q = 0; q < 2; ++q) {
      int group = wave * 2 + q;
      int mt = group & 3, jt = group >> 2;
      int j = jt * 16 + (lane & 15);
      int mrow0 = mt * 16 + laneHi * 8;
      int m = mt * 16 + (lane & 15);

      size_t tb = (size_t)(s * 4 + mt) * 64;   // global row-tile base
      v8f aI = loadCfrag(gxD + ((tb + ( 0 + jt)) * 32 + lane) * 8);
      v8f aF = loadCfrag(gxD + ((tb + (16 + jt)) * 32 + lane) * 8);
      v8f aG = loadCfrag(gxD + ((tb + (32 + jt)) * 32 + lane) * 8);
      v8f aO = loadCfrag(gxD + ((tb + (48 + jt)) * 32 + lane) * 8);

#pragma unroll
      for (int kt = 0; kt < kH / 32; ++kt) {
        v16h a = loadAfragH(hbuf, kH, m, kt * 32 + laneHi * 8);
        v16h b0 = *(const v16h*)(bp + ((size_t)(kt * 64 + ( 0 + jt)) * 32 + lane) * 16);
        v16h b1 = *(const v16h*)(bp + ((size_t)(kt * 64 + (16 + jt)) * 32 + lane) * 16);
        v16h b2 = *(const v16h*)(bp + ((size_t)(kt * 64 + (32 + jt)) * 32 + lane) * 16);
        v16h b3 = *(const v16h*)(bp + ((size_t)(kt * 64 + (48 + jt)) * 32 + lane) * 16);
        aI = wmma_f16(a, b0, aI);
        aF = wmma_f16(a, b1, aF);
        aG = wmma_f16(a, b2, aG);
        aO = wmma_f16(a, b3, aO);
      }
#pragma unroll
      for (int v = 0; v < 8; ++v) {
        float iv = sigm(aI[v]);
        float fv = sigm(aF[v]);
        float gv = tanhf(aG[v]);
        float ov = sigm(aO[v]);
        float c  = fv * cst[q][v] + iv * gv;
        cst[q][v] = c;
        float h = ov * tanhf(c);
        int bb = mrow0 + v;
        size_t o = ((size_t)bb * kS + s) * kHID + dir * kH + j;
        lstm[o]  = h;
        lstmH[o] = (_Float16)h;
        hh[q][v] = (_Float16)h;
      }
    }
    __syncthreads();                 // all reads of old h done
#pragma unroll
    for (int q = 0; q < 2; ++q) {
      int group = wave * 2 + q;
      int mt = group & 3, jt = group >> 2;
      int j = jt * 16 + (lane & 15);
      int mrow0 = mt * 16 + laneHi * 8;
#pragma unroll
      for (int v = 0; v < 8; ++v) hbuf[(mrow0 + v) * kH + j] = hh[q][v];
    }
    __syncthreads();                 // new h visible
  }
}

// ------ 6) scores = tanh(lstmH @ att1T + b1) @ UT  (fused, WMMA) ----------
// att1P staged into LDS per ntb pass (shared by all 8 waves of the block).
__global__ __launch_bounds__(256) void k_scores(const _Float16* __restrict__ lstmH,
                                                const _Float16* __restrict__ att1P,
                                                const float* __restrict__ att1b,
                                                const float* __restrict__ UT,
                                                float* __restrict__ scores) {
  __shared__ _Float16 bsh[16 * 2048];   // 64 KB
  int tid = threadIdx.x;
  int wid = blockIdx.x * 8 + (tid >> 5);
  int lane = tid & 31, laneHi = (lane >> 4) & 1;
  int rowBase = wid * 16;
  int m = rowBase + (lane & 15);
  const _Float16* rowA = lstmH + (size_t)m * kHID;

  float part[8];
#pragma unroll
  for (int v = 0; v < 8; ++v) part[v] = 0.f;

  for (int ntb = 0; ntb < 4; ++ntb) {          // 4 blocks of 4 column tiles
    __syncthreads();                           // previous pass done with bsh
    stage4(bsh, att1P + (size_t)ntb * 2048, 16, 16 * 512, tid);
    asyncWait();
    __syncthreads();

    v8f acc[4];
#pragma unroll
    for (int t = 0; t < 4; ++t) {
      float bv = att1b[(ntb * 4 + t) * 16 + (lane & 15)];
#pragma unroll
      for (int v = 0; v < 8; ++v) acc[t][v] = bv;
    }
    for (int kt = 0; kt < kHID / 32; ++kt) {
      int kb = kt * 32 + laneHi * 8;
      union { v16h v; v8h h[2]; } a;
      a.h[0] = *(const v8h*)(rowA + kb);
      a.h[1] = *(const v8h*)(rowA + kb + 16);
#pragma unroll
      for (int t = 0; t < 4; ++t) {
        v16h b = *(const v16h*)(bsh + ((kt * 4 + t) * 512 + lane * 16));
        acc[t] = wmma_f16(a.v, b, acc[t]);
      }
    }
#pragma unroll
    for (int t = 0; t < 4; ++t) {
      float uv = UT[(ntb * 4 + t) * 16 + (lane & 15)];
#pragma unroll
      for (int v = 0; v < 8; ++v) part[v] += tanhf(acc[t][v]) * uv;
    }
  }
  // butterfly reduce over the 16 lanes holding one row's N values
#pragma unroll
  for (int off = 1; off < 16; off <<= 1)
#pragma unroll
    for (int v = 0; v < 8; ++v) part[v] += __shfl_xor(part[v], off, 32);

  if ((lane & 15) == 0) {
#pragma unroll
    for (int v = 0; v < 8; ++v)
      scores[rowBase + laneHi * 8 + v] = part[v];
  }
}

// ---------------- 7) softmax over time, write alpha (ws + d_out) ----------
__global__ void k_softmax(const float* __restrict__ scores,
                          float* __restrict__ alpha,
                          float* __restrict__ outAlpha) {
  __shared__ float red[256];
  int b = blockIdx.x, t = threadIdx.x;
  float v0 = scores[b * kS + t], v1 = scores[b * kS + 256 + t];
  red[t] = fmaxf(v0, v1);
  __syncthreads();
  for (int off = 128; off > 0; off >>= 1) {
    if (t < off) red[t] = fmaxf(red[t], red[t + off]);
    __syncthreads();
  }
  float mx = red[0];
  __syncthreads();
  float e0 = __expf(v0 - mx), e1 = __expf(v1 - mx);
  red[t] = e0 + e1;
  __syncthreads();
  for (int off = 128; off > 0; off >>= 1) {
    if (t < off) red[t] += red[t + off];
    __syncthreads();
  }
  float inv = 1.f / red[0];
  float a0 = e0 * inv, a1 = e1 * inv;
  alpha[b * kS + t]        = a0;
  alpha[b * kS + 256 + t]  = a1;
  outAlpha[b * kS + t]       = a0;
  outAlpha[b * kS + 256 + t] = a1;
}

// ---------------- 8) attention pooling ----------------
__global__ void k_pool(const float* __restrict__ lstm,
                       const float* __restrict__ alpha,
                       float* __restrict__ pooled) {
  __shared__ float al[kS];
  int b = blockIdx.x, h = threadIdx.x;   // 512 threads
  for (int i = threadIdx.x; i < kS; i += blockDim.x) al[i] = alpha[b * kS + i];
  __syncthreads();
  float acc = 0.f;
  for (int s = 0; s < kS; ++s)
    acc += lstm[((size_t)b * kS + s) * kHID + h] * al[s];
  pooled[b * kHID + h] = acc;
}

// ---------------- 9) classifier head ----------------
__global__ void k_logit(const float* __restrict__ pooled,
                        const float* __restrict__ w2,
                        const float* __restrict__ b2,
                        float* __restrict__ outLogit) {
  int tid = blockIdx.x * blockDim.x + threadIdx.x;
  if (tid >= kB * kC) return;
  int b = tid / kC, c = tid % kC;
  float acc = b2[c];
  for (int k = 0; k < kHID; ++k)
    acc += pooled[b * kHID + k] * w2[c * kHID + k];
  outLogit[b * kC + c] = acc;
}

// ---------------- launch ----------------
extern "C" void kernel_launch(void* const* d_in, const int* in_sizes, int n_in,
                              void* d_out, int out_size, void* d_ws, size_t ws_size,
                              hipStream_t stream) {
  const int*   x    = (const int*)d_in[0];
  const float* emb  = (const float*)d_in[1];
  const float* wihf = (const float*)d_in[2];
  const float* whhf = (const float*)d_in[3];
  const float* bihf = (const float*)d_in[4];
  const float* bhhf = (const float*)d_in[5];
  const float* wihb = (const float*)d_in[6];
  const float* whhb = (const float*)d_in[7];
  const float* bihb = (const float*)d_in[8];
  const float* bhhb = (const float*)d_in[9];
  const float* a1w  = (const float*)d_in[10];
  const float* a1b  = (const float*)d_in[11];
  const float* UT   = (const float*)d_in[12];
  const float* a2w  = (const float*)d_in[13];
  const float* a2b  = (const float*)d_in[14];
  float* out = (float*)d_out;          // [B*C logits | B*S alpha]

  char*  ws  = (char*)d_ws;
  size_t off = 0;
  auto alloc = [&](size_t bytes) -> char* {
    char* p = ws + off;
    off = (off + bytes + 255) & ~(size_t)255;
    return p;
  };
  _Float16* embH  = (_Float16*)alloc((size_t)kS * kB * kE * 2);          // 33.5 MB
  float*    gxP   = (float*)   alloc((size_t)2 * kS * kB * kG * 4);      // 268 MB
  _Float16* wihP  = (_Float16*)alloc((size_t)2 * kE * kG * 2);           // 2 MB
  _Float16* whhP  = (_Float16*)alloc((size_t)2 * kH * kG * 2);           // 1 MB
  float*    biasC = (float*)   alloc((size_t)2 * kG * 4);
  _Float16* att1P = (_Float16*)alloc((size_t)kHID * kATT * 2);
  float*    lstm  = (float*)   alloc((size_t)kB * kS * kHID * 4);        // 67 MB
  _Float16* lstmH = (_Float16*)alloc((size_t)kB * kS * kHID * 2);        // 33.5 MB
  float*    scor  = (float*)   alloc((size_t)kB * kS * 4);
  float*    alpha = (float*)   alloc((size_t)kB * kS * 4);
  float*    pooled= (float*)   alloc((size_t)kB * kHID * 4);
  if (off > ws_size) return;   // workspace too small: nothing safe to do

  // 1) gather to f16 time-major
  k_gather<<<(kS * kB * kE / 8) / 256, 256, 0, stream>>>(x, emb, embH);

  // 2) pack weights into WMMA-B fragment order
  k_pack<<<(kE * kG) / 256, 256, 0, stream>>>(wihf, wihP,              kE, kG / 16);
  k_pack<<<(kE * kG) / 256, 256, 0, stream>>>(wihb, wihP + (size_t)kE * kG, kE, kG / 16);
  k_pack<<<(kH * kG) / 256, 256, 0, stream>>>(whhf, whhP,              kH, kG / 16);
  k_pack<<<(kH * kG) / 256, 256, 0, stream>>>(whhb, whhP + (size_t)kH * kG, kH, kG / 16);
  k_pack<<<(kHID * kATT) / 256, 256, 0, stream>>>(a1w, att1P, kHID, kATT / 16);

  // 3) combined biases
  k_bias<<<kG / 256, 256, 0, stream>>>(bihf, bhhf, biasC);
  k_bias<<<kG / 256, 256, 0, stream>>>(bihb, bhhb, biasC + kG);

  // 4) big input-projection GEMM: block = 8 row tiles, B staged in LDS
  {
    dim3 g(kS * kB / 16 / 8, kG / 64, 2);
    k_gemm_gx<<<g, 256, 0, stream>>>(embH, wihP, biasC, gxP);
  }

  // 5) sequential recurrence: 1 persistent workgroup per direction
  k_lstm<<<2, 1024, 0, stream>>>(gxP, whhP, lstm, lstmH);

  // 6) attention scores
  k_scores<<<(kB * kS / 16) / 8, 256, 0, stream>>>(lstmH, att1P, a1b, UT, scor);

  // 7) softmax over time -> alpha (also second output)
  k_softmax<<<kB, 256, 0, stream>>>(scor, alpha, out + kB * kC);

  // 8) attention pooling
  k_pool<<<kB, kHID, 0, stream>>>(lstm, alpha, pooled);

  // 9) logits (first output)
  k_logit<<<(kB * kC + 255) / 256, 256, 0, stream>>>(pooled, a2w, a2b, out);
}
